// equivariant_join_71665824301382
// MI455X (gfx1250) — matrix operand order repaired and checked by reading, not compiled, not run on hardware
//
#include <hip/hip_runtime.h>

// MI455X / gfx1250, wave32. Strategy:
//   out[n,i] = ref[n,31] * sum_{j,k} W3[i,j,k] * x[n,j] * y[n,k]
// where W3 folds outer/signs/permutation (values in {-1,0,1}, exact in f16).
// GEMM via v_wmma_f32_16x16x32_f16: M=items(16), N=out-blades(16 x2), K=1024
// (32 chunks of 32; chunk j covers k=0..31). A built on the fly as x[j]*y[k]
// (v_pk_mul_f16, co-executes with XDL), B pre-swizzled into WMMA register
// order by a tiny prep kernel into d_ws (64KB, L2-resident).

typedef __attribute__((ext_vector_type(16))) _Float16 v16h;
typedef __attribute__((ext_vector_type(8)))  float    v8f;

#define TILES_PER_WAVE 4   // 16-item tiles per wave (64 items/wave)

// ---------------------------------------------------------------------------
// Prep: Bbuf[((j*2+h)*32 + lane)*16 + e] = W3[i][j][k] as f16, laid out so the
// main kernel's v16h load per lane is exactly the WMMA B operand for
// 16x16x32: lane<16 -> column n=lane, K=e(0..15); lane>=16 -> n=lane-16,
// K=16+e. W3[i][j][k] = signs[i]*signs[ipj]*signs[ipk]*outer[perm[i]][ipj][ipk]
// with ip = inverse permutation (coefficient extraction of x[j], y[k]).
// ---------------------------------------------------------------------------
__global__ __launch_bounds__(256) void gj_prep_kernel(
    const float* __restrict__ outer, const float* __restrict__ signs,
    const int* __restrict__ perm, _Float16* __restrict__ Bbuf)
{
  int t = blockIdx.x * blockDim.x + threadIdx.x;
  if (t >= 32 * 2 * 32 * 16) return;
  int e    =  t        & 15;
  int lane = (t >> 4)  & 31;
  int h    = (t >> 9)  & 1;
  int j    =  t >> 10;
  int i = h * 16 + (lane & 15);
  int k = (lane < 16) ? e : (16 + e);
  int ipj = 0, ipk = 0;
  for (int q = 0; q < 32; ++q) {
    if (perm[q] == j) ipj = q;
    if (perm[q] == k) ipk = q;
  }
  int pi = perm[i];
  float w = signs[i] * signs[ipj] * signs[ipk] *
            outer[((size_t)pi * 32 + ipj) * 32 + ipk];
  Bbuf[t] = (_Float16)w;
}

// ---------------------------------------------------------------------------
// Main: each wave handles TILES_PER_WAVE tiles of 16 items. K-chunk loop is
// outermost so the 64KB B operand is streamed once per 64 items (hot in L0/L2).
// A-operand per chunk j: lane's 16 f16 values = x[item][j] * ysel, where ysel
// uses the contiguous K order of Bbuf (lane<16: k = 0..15;
// lane>=16: k = 16..31).
// ---------------------------------------------------------------------------
__global__ __launch_bounds__(256) void gj_main_kernel(
    const float* __restrict__ x, const float* __restrict__ y,
    const float* __restrict__ ref, const _Float16* __restrict__ Bbuf,
    float* __restrict__ out, int nItems)
{
  const int lane = threadIdx.x & 31;
  const int wave = blockIdx.x * 8 + (threadIdx.x >> 5);
  const int itemBase = wave * (16 * TILES_PER_WAVE);
  if (itemBase + 16 * TILES_PER_WAVE > nItems) return;  // wave-uniform

  const bool hi = (lane >= 16);
  const int  m  = lane & 15;

  v16h xh[TILES_PER_WAVE][2];   // f16 x row, j = vec*16 + elem
  v16h ysel[TILES_PER_WAVE];    // f16 y values in A-layout K order
  v8f  acc0[TILES_PER_WAVE], acc1[TILES_PER_WAVE];

  #pragma unroll
  for (int t = 0; t < TILES_PER_WAVE; ++t) {
    const int item = itemBase + t * 16 + m;
    const float4* xr = (const float4*)(x + (size_t)item * 32);
    const float4* yr = (const float4*)(y + (size_t)item * 32);
    float yf[32];
    #pragma unroll
    for (int q = 0; q < 8; ++q) {
      float4 xv = xr[q];
      xh[t][q >> 2][(q & 3) * 4 + 0] = (_Float16)xv.x;
      xh[t][q >> 2][(q & 3) * 4 + 1] = (_Float16)xv.y;
      xh[t][q >> 2][(q & 3) * 4 + 2] = (_Float16)xv.z;
      xh[t][q >> 2][(q & 3) * 4 + 3] = (_Float16)xv.w;
      float4 yv = yr[q];
      yf[q * 4 + 0] = yv.x; yf[q * 4 + 1] = yv.y;
      yf[q * 4 + 2] = yv.z; yf[q * 4 + 3] = yv.w;
    }
    #pragma unroll
    for (int e = 0; e < 8; ++e) {
      ysel[t][e]     = (_Float16)(hi ? yf[16 + e] : yf[e]);
      ysel[t][e + 8] = (_Float16)(hi ? yf[24 + e] : yf[8 + e]);
    }
    v8f z = {};
    acc0[t] = z;
    acc1[t] = z;
  }

  #pragma unroll
  for (int j = 0; j < 32; ++j) {
    const v16h b0 = *(const v16h*)(Bbuf + (((size_t)j * 2 + 0) * 32 + lane) * 16);
    const v16h b1 = *(const v16h*)(Bbuf + (((size_t)j * 2 + 1) * 32 + lane) * 16);
    #pragma unroll
    for (int t = 0; t < TILES_PER_WAVE; ++t) {
      const _Float16 xj = xh[t][j >> 4][j & 15];
      const v16h a = ysel[t] * xj;  // 8x v_pk_mul_f16, co-executes with XDL
      acc0[t] = __builtin_amdgcn_wmma_f32_16x16x32_f16(
          false, a, false, b0, (short)0, acc0[t], false, false);
      acc1[t] = __builtin_amdgcn_wmma_f32_16x16x32_f16(
          false, a, false, b1, (short)0, acc1[t], false, false);
    }
  }

  // D layout: VGPR r -> item r (lanes 0-15) / item r+8 (lanes 16-31),
  // column i = (lane&15) + 16*tilehalf. Scale by ref_ps and store (64B segs).
  #pragma unroll
  for (int t = 0; t < TILES_PER_WAVE; ++t) {
    #pragma unroll
    for (int r = 0; r < 8; ++r) {
      const int item = itemBase + t * 16 + r + (hi ? 8 : 0);
      const float rp = ref[(size_t)item * 32 + 31];
      out[(size_t)item * 32 + m]      = acc0[t][r] * rp;
      out[(size_t)item * 32 + 16 + m] = acc1[t][r] * rp;
    }
  }
}

extern "C" void kernel_launch(void* const* d_in, const int* in_sizes, int n_in,
                              void* d_out, int out_size, void* d_ws, size_t ws_size,
                              hipStream_t stream) {
  const float* x      = (const float*)d_in[0];
  const float* y      = (const float*)d_in[1];
  const float* ref    = (const float*)d_in[2];
  const float* outer  = (const float*)d_in[3];
  const float* signs  = (const float*)d_in[4];
  const int*   perm   = (const int*)d_in[5];
  float*       out    = (float*)d_out;
  _Float16*    Bbuf   = (_Float16*)d_ws;   // 64 KB swizzled f16 B operand

  const int nItems = in_sizes[0] / 32;     // 1048576

  // 32*2*32*16 = 32768 B elements
  gj_prep_kernel<<<128, 256, 0, stream>>>(outer, signs, perm, Bbuf);

  const int itemsPerWave = 16 * TILES_PER_WAVE;          // 64
  const int waves  = nItems / itemsPerWave;              // 16384
  const int blocks = (waves + 7) / 8;                    // 2048 (8 waves/block)
  gj_main_kernel<<<blocks, 256, 0, stream>>>(x, y, ref, Bbuf, out, nItems);
}
